// FullAttention_9947144258109
// MI455X (gfx1250) — matrix operand order repaired and checked by reading, not compiled, not run on hardware
//
#include <hip/hip_runtime.h>

// ---------------------------------------------------------------------------
// Fused MHA block for MI455X (gfx1250, wave32, WMMA).
// B=4, S=2048, D=512, H=8, DK=DV=64.  All GEMMs on v_wmma_f32_16x16x32_f16.
// ---------------------------------------------------------------------------

typedef __attribute__((ext_vector_type(16))) _Float16 v16h;
typedef __attribute__((ext_vector_type(8)))  _Float16 v8h;
typedef __attribute__((ext_vector_type(8)))  float    v8f;
typedef __attribute__((ext_vector_type(4)))  float    v4f;

union V16U { v16h v; v8h h[2]; };

#define B_  4
#define S_  2048
#define D_  512
#define H_  8
#define DK_ 64
#define DV_ 64
#define HD_ 512   // H*DK == H*DV

__device__ __forceinline__ v8f wmma16(v16h a, v16h b, v8f c) {
  // D = A(16x32) * B(32x16) + C(16x16), f32 accumulate
  return __builtin_amdgcn_wmma_f32_16x16x32_f16(
      /*neg_a=*/false, a, /*neg_b=*/false, b,
      /*c_mod=*/(short)0, c, /*reuse_a=*/false, /*reuse_b=*/false);
}

// ---------------------------------------------------------------------------
// Kernel 1: one projection GEMM (templated: 0->Q, 1->K, 2->V).
// One wave computes one 16x16 output tile; 16 WMMAs along K=512.
//   Q,K stored f16 [B,H,S,DK] (dk contiguous)  -- scale 1/8 folded into Q
//   V   stored f16 [B,H,DV,S] (s  contiguous)  -- transposed for B-matrix use
// ---------------------------------------------------------------------------
template <int WHICH>
__global__ __launch_bounds__(32)
void proj_kernel(const float* __restrict__ X, const float* __restrict__ W,
                 _Float16* __restrict__ Out)
{
  const int lane = threadIdx.x & 31;
  const int hi   = lane >> 4;      // half-wave: 0 or 1
  const int ln   = lane & 15;
  const int n0 = blockIdx.x * 16;
  const int m0 = blockIdx.y * 16;

  const int row  = m0 + ln;        // A-matrix: lane -> M
  const int ncol = n0 + ln;        // B-matrix: lane -> N

  v8f c = {};
  for (int kc = 0; kc < D_; kc += 32) {
    // A tile 16x32 f16, contiguous k order: half group g (v8h) -> K = hi*16 + g*8 + i
    V16U a;
#pragma unroll
    for (int g = 0; g < 2; ++g) {
      const int kb = kc + hi * 16 + g * 8;
      const v4f x0 = *(const v4f*)(X + (size_t)row * D_ + kb);
      const v4f x1 = *(const v4f*)(X + (size_t)row * D_ + kb + 4);
      v8h t;
#pragma unroll
      for (int i = 0; i < 4; ++i) { t[i] = (_Float16)x0[i]; t[4 + i] = (_Float16)x1[i]; }
      a.h[g] = t;
    }
    // B tile 32x16 f16, same contiguous k order as A: lane -> N, half h -> K = hi*16 + h
    V16U b;
    const int kb = kc + hi * 16;
#pragma unroll
    for (int t = 0; t < 16; ++t)
      b.v[t] = (_Float16)W[(size_t)(kb + t) * HD_ + ncol];
    c = wmma16(a.v, b.v, c);
  }

  // C layout: VGPR r, lanes0-15 -> M=r, lanes16-31 -> M=r+8
  const int hh = ncol >> 6;        // head
  const int dk = ncol & 63;        // dim within head
#pragma unroll
  for (int r = 0; r < 8; ++r) {
    const int m  = m0 + r + hi * 8;
    const int bb = m >> 11;          // m / S_
    const int s  = m & (S_ - 1);
    if (WHICH == 0) {
      Out[(((size_t)(bb * H_ + hh) * S_) + s) * DK_ + dk] = (_Float16)(c[r] * 0.125f);
    } else if (WHICH == 1) {
      Out[(((size_t)(bb * H_ + hh) * S_) + s) * DK_ + dk] = (_Float16)c[r];
    } else {
      Out[(((size_t)(bb * H_ + hh) * DV_) + dk) * S_ + s]  = (_Float16)c[r];
    }
  }
}

// ---------------------------------------------------------------------------
// Kernel 2: flash-style attention.  One wave per (b, h, 16 query rows).
// Per 32-key step: 4 WMMA (scores) + mask + online softmax + 4 WMMA (P*V).
// ---------------------------------------------------------------------------
__global__ __launch_bounds__(32)
void attention_kernel(const _Float16* __restrict__ Qh, const _Float16* __restrict__ Kh,
                      const _Float16* __restrict__ Vt, const unsigned char* __restrict__ mask,
                      _Float16* __restrict__ Ctx)
{
  // P tile staging (C-layout -> A-layout). Row = 40 halves = 80B (16B multiple)
  // so the A-layout read-back is two ds_load_b128 per lane.
  __shared__ _Float16 pl[16][40];

  const int lane = threadIdx.x & 31;
  const int hi   = lane >> 4;
  const int ln   = lane & 15;
  const int q0 = blockIdx.x * 16;
  const int h  = blockIdx.y;
  const int b  = blockIdx.z;

  const _Float16* Qb = Qh + (size_t)(b * H_ + h) * S_ * DK_;
  const _Float16* Kb = Kh + (size_t)(b * H_ + h) * S_ * DK_;
  const _Float16* Vb = Vt + (size_t)(b * H_ + h) * DV_ * S_;

  // Q A-tiles for dk-chunks {0,32} (kept in registers for the whole pass)
  V16U aq[2];
#pragma unroll
  for (int kc = 0; kc < 2; ++kc)
#pragma unroll
    for (int g = 0; g < 2; ++g) {
      const int dkb = kc * 32 + hi * 16 + g * 8;
      aq[kc].h[g] = *(const v8h*)(Qb + (size_t)(q0 + ln) * DK_ + dkb);
    }

  float mrun[8], lrun[8];
  v8f acc[4] = {};
#pragma unroll
  for (int r = 0; r < 8; ++r) { mrun[r] = -INFINITY; lrun[r] = 0.0f; }

  const int qrow = q0 + hi * 8;

  for (int j = 0; j < S_; j += 32) {
    // ---- scores: two 16x16 tiles (keys j..j+15, j+16..j+31) ----
    v8f sc[2] = {};
#pragma unroll
    for (int jt = 0; jt < 2; ++jt) {
      const int key = j + jt * 16 + ln;
#pragma unroll
      for (int kc = 0; kc < 2; ++kc) {
        // B tile: 16 contiguous halves (32B, aligned) -> global_load_b128 x2
        const v16h bk = *(const v16h*)(Kb + (size_t)key * DK_ + kc * 32 + hi * 16);
        sc[jt] = wmma16(aq[kc].v, bk, sc[jt]);
      }
    }

    // ---- mask (true -> -inf in reference; -1e30 keeps online max finite) ----
#pragma unroll
    for (int jt = 0; jt < 2; ++jt) {
      const int key = j + jt * 16 + ln;
#pragma unroll
      for (int r = 0; r < 8; ++r) {
        const int q = qrow + r;
        if (mask[((size_t)(b * S_ + q)) * S_ + key]) sc[jt][r] = -1e30f;
      }
    }

    // ---- online softmax: a row lives in one VGPR across one half-wave ----
    float alpha[8];
#pragma unroll
    for (int r = 0; r < 8; ++r) {
      float rm = fmaxf(sc[0][r], sc[1][r]);
#pragma unroll
      for (int x = 1; x < 16; x <<= 1) rm = fmaxf(rm, __shfl_xor(rm, x, 32));
      const float mnew = fmaxf(mrun[r], rm);
      alpha[r] = __expf(mrun[r] - mnew);
      mrun[r] = mnew;
    }

    v8f p[2];
#pragma unroll
    for (int r = 0; r < 8; ++r) {
      p[0][r] = __expf(sc[0][r] - mrun[r]);
      p[1][r] = __expf(sc[1][r] - mrun[r]);
      float rs = p[0][r] + p[1][r];
#pragma unroll
      for (int x = 1; x < 16; x <<= 1) rs += __shfl_xor(rs, x, 32);
      lrun[r] = lrun[r] * alpha[r] + rs;
#pragma unroll
      for (int nt = 0; nt < 4; ++nt) acc[nt][r] *= alpha[r];
    }

    // ---- relayout P: C-matrix layout -> A-matrix layout via LDS ----
#pragma unroll
    for (int jt = 0; jt < 2; ++jt)
#pragma unroll
      for (int r = 0; r < 8; ++r)
        pl[r + hi * 8][jt * 16 + ln] = (_Float16)p[jt][r];
    asm volatile("s_wait_dscnt 0" ::: "memory");   // single-wave LDS RAW
    V16U pa;
#pragma unroll
    for (int g = 0; g < 2; ++g)
      pa.h[g] = *(const v8h*)&pl[ln][hi * 16 + g * 8];

    // ---- context += P(16x32) * V(32x64): 4 WMMAs ----
#pragma unroll
    for (int nt = 0; nt < 4; ++nt) {
      const int dv = nt * 16 + ln;
      const v16h bv = *(const v16h*)(Vb + (size_t)dv * S_ + j + hi * 16);
      acc[nt] = wmma16(pa.v, bv, acc[nt]);
    }

    // prefetch next K tile (emits global_prefetch_b8)
    if (j + 32 < S_) {
      __builtin_prefetch(Kb + (size_t)(j + 32 + ln) * DK_, 0, 1);
      __builtin_prefetch(Kb + (size_t)(j + 48 + ln) * DK_, 0, 1);
    }
  }

  // ---- epilogue: normalize rows, store context f16 [B, S, H*DV] ----
#pragma unroll
  for (int r = 0; r < 8; ++r) {
    const int q = qrow + r;
    const float inv = 1.0f / lrun[r];
#pragma unroll
    for (int nt = 0; nt < 4; ++nt) {
      const int col = h * DV_ + nt * 16 + ln;
      Ctx[((size_t)(b * S_ + q)) * HD_ + col] = (_Float16)(acc[nt][r] * inv);
    }
  }
}

// ---------------------------------------------------------------------------
// Kernel 3: out = LN(Ctx @ W_fc + input_Q).  256 threads = 8 waves per
// 16-row block; each wave owns 64 output columns (4 WMMA tiles, 64 WMMAs).
// Row staged in LDS f32; per-row mean/var via half-wave shuffles.
// ---------------------------------------------------------------------------
__global__ __launch_bounds__(256)
void fc_ln_kernel(const _Float16* __restrict__ Ctx, const float* __restrict__ Wfc,
                  const float* __restrict__ Xq, const float* __restrict__ gamma,
                  const float* __restrict__ beta, float* __restrict__ out)
{
  __shared__ float xb[16][D_ + 4];

  const int tid  = threadIdx.x;
  const int lane = tid & 31;
  const int wave = tid >> 5;
  const int hi   = lane >> 4;
  const int ln   = lane & 15;
  const int m0 = blockIdx.x * 16;
  const int row = m0 + ln;

  v8f c[4] = {};
  for (int kc = 0; kc < D_; kc += 32) {
    V16U a;
#pragma unroll
    for (int g = 0; g < 2; ++g)
      a.h[g] = *(const v8h*)(Ctx + (size_t)row * HD_ + kc + hi * 16 + g * 8);
#pragma unroll
    for (int nt = 0; nt < 4; ++nt) {
      const int ncol = wave * 64 + nt * 16 + ln;
      V16U b;
      const int kb = kc + hi * 16;
#pragma unroll
      for (int t = 0; t < 16; ++t)
        b.v[t] = (_Float16)Wfc[(size_t)(kb + t) * D_ + ncol];
      c[nt] = wmma16(a.v, b.v, c[nt]);
    }
  }

  // residual add, stage f32 row tile in LDS
#pragma unroll
  for (int nt = 0; nt < 4; ++nt) {
#pragma unroll
    for (int r = 0; r < 8; ++r) {
      const int m    = m0 + r + hi * 8;
      const int ncol = wave * 64 + nt * 16 + ln;
      xb[r + hi * 8][ncol] = c[nt][r] + Xq[(size_t)m * D_ + ncol];
    }
  }
  __syncthreads();

  // LayerNorm: 16 threads per row (contiguous 16 lanes of one half-wave)
  const int rrow = tid >> 4;
  const int sub  = tid & 15;
  float sum = 0.0f, sq = 0.0f;
#pragma unroll
  for (int i = 0; i < 32; ++i) {
    const float v = xb[rrow][sub * 32 + i];
    sum += v; sq += v * v;
  }
#pragma unroll
  for (int x = 1; x < 16; x <<= 1) {
    sum += __shfl_xor(sum, x, 32);
    sq  += __shfl_xor(sq,  x, 32);
  }
  const float mean = sum * (1.0f / D_);
  const float var  = sq * (1.0f / D_) - mean * mean;
  const float rstd = rsqrtf(var + 1e-5f);
  const int gm = m0 + rrow;
#pragma unroll
  for (int i = 0; i < 32; ++i) {
    const int colc = sub * 32 + i;
    out[(size_t)gm * D_ + colc] = (xb[rrow][colc] - mean) * rstd * gamma[colc] + beta[colc];
  }
}

// ---------------------------------------------------------------------------
// Launch.  Workspace layout (f16): Qh | Kh | Vt | Ctx, 4M elements each (32MB).
// ---------------------------------------------------------------------------
extern "C" void kernel_launch(void* const* d_in, const int* in_sizes, int n_in,
                              void* d_out, int out_size, void* d_ws, size_t ws_size,
                              hipStream_t stream)
{
  (void)in_sizes; (void)n_in; (void)out_size; (void)ws_size;

  const float* Xq   = (const float*)d_in[0];
  const float* Xk   = (const float*)d_in[1];
  const float* Xv   = (const float*)d_in[2];
  const unsigned char* mask = (const unsigned char*)d_in[3];  // jnp.bool_ -> 1 byte
  const float* Wq   = (const float*)d_in[4];
  const float* Wk   = (const float*)d_in[5];
  const float* Wv   = (const float*)d_in[6];
  const float* Wfc  = (const float*)d_in[7];
  const float* gam  = (const float*)d_in[8];
  const float* bet  = (const float*)d_in[9];
  float* out = (float*)d_out;

  _Float16* ws = (_Float16*)d_ws;
  const size_t mat = (size_t)B_ * S_ * HD_;   // 4M elements
  _Float16* Qh  = ws;
  _Float16* Kh  = ws + mat;
  _Float16* Vt  = ws + 2 * mat;
  _Float16* Ctx = ws + 3 * mat;

  dim3 g1(HD_ / 16, (B_ * S_) / 16);
  proj_kernel<0><<<g1, 32, 0, stream>>>(Xq, Wq, Qh);
  proj_kernel<1><<<g1, 32, 0, stream>>>(Xk, Wk, Kh);
  proj_kernel<2><<<g1, 32, 0, stream>>>(Xv, Wv, Vt);

  dim3 g2(S_ / 16, H_, B_);
  attention_kernel<<<g2, 32, 0, stream>>>(Qh, Kh, Vt, mask, Ctx);

  fc_ln_kernel<<<(B_ * S_) / 16, 256, 0, stream>>>(Ctx, Wfc, Xq, gam, bet, out);
}
